// MoEPositionwiseFeedForward_1649267442344
// MI455X (gfx1250) — matrix operand order repaired and checked by reading, not compiled, not run on hardware
//
#include <hip/hip_runtime.h>
#include <math.h>

// ---------------------------------------------------------------------------
// MoE positionwise FFN for MI455X (gfx1250).
// Expert GEMMs use V_WMMA_F32_16X16X32_BF16 with a bf16x2 split / 3-product
// scheme (ah*bh + ah*bl + al*bh, f32 accumulate): ~1e-4 relative accuracy
// (fp32-grade for tolerance purposes) at the chip's native 16-bit matrix
// width, ~2.7x fewer matrix-pipe issues than the f32 16x16x4 path. The
// split/convert VALU work co-executes with the WMMA (XDL 16-bit ops are
// TRANS-tracked). Working set (x + w1 + w2 ~ 167MB) is L2-resident (192MB).
// ---------------------------------------------------------------------------

#define T_TOK 32768
#define D_DIM 512
#define F_DIM 1024
#define E_NUM 16
#define CAP   2048

typedef __attribute__((ext_vector_type(16))) __bf16 v16bf;
typedef __attribute__((ext_vector_type(8)))  float  v8f;

__device__ __forceinline__ v8f wmma_bf16(v16bf a, v16bf b, v8f c) {
  return __builtin_amdgcn_wmma_f32_16x16x32_bf16(false, a, false, b,
                                                 (short)0, c, false, false);
}

// Split 16 f32 values into bf16 hi + bf16 lo (residual) fragments.
__device__ __forceinline__ void split_frag(const float* __restrict__ v,
                                           v16bf& hi, v16bf& lo) {
#pragma unroll
  for (int j = 0; j < 16; ++j) {
    float x = v[j];
    __bf16 h = (__bf16)x;
    hi[j] = h;
    lo[j] = (__bf16)(x - (float)h);
  }
}

// ---------------------------------------------------------------- zero init
__global__ __launch_bounds__(256) void moe_zero_kernel(float* __restrict__ out,
                                                       int out_n,
                                                       float* __restrict__ esum) {
  int i = blockIdx.x * 256 + threadIdx.x;
  if (i < out_n) out[i] = 0.0f;
  if (i < E_NUM) esum[i] = 0.0f;
}

// ---------------------------------------------------------------- gating
// block: 256 threads = 16 tokens x 16 experts. Exact softmax + top-2.
__global__ __launch_bounds__(256) void moe_gate_kernel(
    const float* __restrict__ x,     // [T, D]
    const float* __restrict__ wg,    // [D, E]
    float* __restrict__ w_et,        // [E, T]
    float* __restrict__ esum) {      // [E] global score sums
  __shared__ float xs[16][516];      // 516 % 64 == 4 -> conflict-free rows
  __shared__ float ssum[16];
  const int tid = threadIdx.x;
  const int t0 = blockIdx.x * 16;

  for (int row = 0; row < 16; ++row) {
    const float* src = x + (size_t)(t0 + row) * D_DIM;
    for (int k = tid; k < D_DIM; k += 256) xs[row][k] = src[k];
  }
  if (tid < 16) ssum[tid] = 0.0f;
  __syncthreads();

  const int tokL = tid >> 4;   // 0..15
  const int e    = tid & 15;   // expert id == lane % 16

  float acc = 0.0f;
  for (int k = 0; k < D_DIM; ++k) acc += xs[tokL][k] * wg[k * E_NUM + e];

  // softmax over the 16-lane group (xor masks <= 8 stay inside the half-wave)
  float m = acc;
  for (int o = 1; o < 16; o <<= 1) m = fmaxf(m, __shfl_xor(m, o));
  float ex = expf(acc - m);
  float s = ex;
  for (int o = 1; o < 16; o <<= 1) s += __shfl_xor(s, o);
  float score = ex / s;

  // top-1 (tie -> lowest index, matching lax.top_k)
  float v1 = score; int i1 = e;
  for (int o = 1; o < 16; o <<= 1) {
    float ov = __shfl_xor(v1, o); int oi = __shfl_xor(i1, o);
    if (ov > v1 || (ov == v1 && oi < i1)) { v1 = ov; i1 = oi; }
  }
  // top-2 (exclude argmax lane)
  float v2 = (e == i1) ? -1.0f : score;
  int   i2 = (e == i1) ? 1000  : e;
  for (int o = 1; o < 16; o <<= 1) {
    float ov = __shfl_xor(v2, o); int oi = __shfl_xor(i2, o);
    if (ov > v2 || (ov == v2 && oi < i2)) { v2 = ov; i2 = oi; }
  }

  float w = (e == i1) ? v1 : ((e == i2) ? v2 : 0.0f);
  const int tok = t0 + tokL;
  w_et[(size_t)e * T_TOK + tok] = w;

  atomicAdd(&ssum[e], score);
  __syncthreads();
  if (tid < 16) atomicAdd(&esum[tid], ssum[tid]);
}

// ---------------------------------------------------------------- aux loss
__global__ void moe_aux_kernel(const float* __restrict__ esum,
                               float* __restrict__ out_aux) {
  if (threadIdx.x == 0) {
    float a = 0.0f;
    for (int e = 0; e < E_NUM; ++e) {
      float mean = esum[e] / (float)T_TOK;
      a += mean * mean;
    }
    out_aux[0] = (float)E_NUM * a;
  }
}

// ---------------------------------------------------------------- selection
// Exact top-CAP per expert via 4-pass MSD radix select on float bits
// (all w_et >= 0, so uint bit order == float order).
__global__ __launch_bounds__(256) void moe_select_kernel(
    const float* __restrict__ w_et,  // [E, T]
    int* __restrict__ sel_idx,       // [E, CAP]
    float* __restrict__ sel_w) {     // [E, CAP]
  __shared__ unsigned hist[256];
  __shared__ unsigned s_prefix;
  __shared__ int s_remaining;
  __shared__ unsigned s_cnt_gt, s_cnt_eq;

  const int e = blockIdx.x;
  const int tid = threadIdx.x;
  const float* w = w_et + (size_t)e * T_TOK;

  if (tid == 0) { s_prefix = 0u; s_remaining = CAP; }
  __syncthreads();

  for (int shift = 24; shift >= 0; shift -= 8) {
    hist[tid] = 0u;
    __syncthreads();
    const unsigned prefix = s_prefix;
    const unsigned hmask = (shift == 24) ? 0u : (0xFFFFFFFFu << (shift + 8));
    for (int t = tid; t < T_TOK; t += 256) {
      unsigned key = __float_as_uint(w[t]);
      if ((key & hmask) == prefix)
        atomicAdd(&hist[(key >> shift) & 255u], 1u);
    }
    __syncthreads();
    if (tid == 0) {
      int cum = 0;
      const int rem = s_remaining;
      for (int d = 255; d >= 0; --d) {
        int c = (int)hist[d];
        if (cum + c >= rem) {
          s_prefix = prefix | ((unsigned)d << shift);
          s_remaining = rem - cum;   // slots still needed at/below this digit
          break;
        }
        cum += c;
      }
    }
    __syncthreads();
  }

  const unsigned thr = s_prefix;
  const int slots_eq = s_remaining;        // slots filled by keys == thr
  const int base_eq = CAP - slots_eq;      // keys > thr fill [0, base_eq)
  if (tid == 0) { s_cnt_gt = 0u; s_cnt_eq = 0u; }
  __syncthreads();

  for (int t = tid; t < T_TOK; t += 256) {
    float wv = w[t];
    unsigned key = __float_as_uint(wv);
    if (key > thr) {
      unsigned pos = atomicAdd(&s_cnt_gt, 1u);
      sel_idx[e * CAP + pos] = t;
      sel_w[e * CAP + pos] = wv;
    } else if (key == thr) {
      unsigned pos = atomicAdd(&s_cnt_eq, 1u);
      if (pos < (unsigned)slots_eq) {
        sel_idx[e * CAP + base_eq + (int)pos] = t;
        sel_w[e * CAP + base_eq + (int)pos] = wv;
      }
    }
  }
}

// ---------------------------------------------------------------- experts
// grid = (CAP/16, E). 256 threads = 8 waves per block.
// Per block: stage 16 gathered token rows (K=512, f32) in LDS; for each
// 128-wide F chunk each wave computes one 16x16 h1 tile + one 16x16 h2 tile
// via bf16x2-split WMMA (K in chunks of 32, 3 products each), applies SwiGLU
// into an LDS act chunk, then all waves accumulate GEMM2 partials (4 N-tiles
// each => 64 output cols). Finally scale by sel_w and scatter-add.
//
// bf16 fragment element -> matrix coordinate maps used here (contiguous K, same for A and B):
//   A 16x32:  elem j of lane(half,r): M=r, K = 16*half + j
//   B 32x16:  elem j of lane(half,r): N=r, K = 16*half + j
//   C/D:      vgpr i of lane(half,r): M = i + 8*half, N = r
__global__ __launch_bounds__(256) void moe_expert_kernel(
    const float* __restrict__ x,       // [T, D]
    const float* __restrict__ w1,      // [E, D, 2F]
    const float* __restrict__ w2,      // [E, F, D]
    const int* __restrict__ sel_idx,   // [E, CAP]
    const float* __restrict__ sel_w,   // [E, CAP]
    float* __restrict__ out) {         // [T, D]
  __shared__ float Alds[16][516];      // gathered x rows, stride%64==4
  __shared__ float ActC[16][132];      // act chunk (K slice of 128)

  const int e = blockIdx.y;
  const int m0 = blockIdx.x * 16;
  const int tid = threadIdx.x;
  const int wave = tid >> 5;
  const int lane = tid & 31;
  const int half = lane >> 4;
  const int r = lane & 15;

  // stage gathered A rows (f32)
  for (int row = 0; row < 16; ++row) {
    const int tok = sel_idx[e * CAP + m0 + row];
    const float* src = x + (size_t)tok * D_DIM;
    for (int k = tid; k < D_DIM; k += 256) Alds[row][k] = src[k];
  }
  __syncthreads();

  const float* w1e = w1 + (size_t)e * D_DIM * (2 * F_DIM);
  const float* w2e = w2 + (size_t)e * F_DIM * D_DIM;
  const int n0 = wave * 64;  // this wave's 64 output columns (of D=512)

  v8f oacc[4] = {{}, {}, {}, {}};

  for (int fc = 0; fc < F_DIM; fc += 128) {
    const int col = fc + wave * 16;    // h1 column base; h2 base = col + F_DIM
    // ---- GEMM1: h1/h2 16x16 tiles, K = 512 in chunks of 32 ----
    v8f acc1 = {}, acc2 = {};
    for (int kc = 0; kc < D_DIM; kc += 32) {
      // A fragment: 16 f32 from LDS -> bf16 hi/lo
      float av[16];
      const int ka = kc + 16 * half;
#pragma unroll
      for (int j = 0; j < 8; ++j) av[j] = Alds[r][ka + j];
#pragma unroll
      for (int j = 0; j < 8; ++j) av[8 + j] = Alds[r][ka + 8 + j];
      v16bf ah, al;
      split_frag(av, ah, al);

      // B fragments for h1 and h2 columns (strided f32 loads, L2-resident)
      const int kb = kc + 16 * half;
      const float* bp = w1e + (size_t)kb * (2 * F_DIM) + col + r;
      float bv1[16], bv2[16];
#pragma unroll
      for (int j = 0; j < 16; ++j) {
        bv1[j] = bp[(size_t)j * (2 * F_DIM)];
        bv2[j] = bp[(size_t)j * (2 * F_DIM) + F_DIM];
      }
      v16bf b1h, b1l, b2h, b2l;
      split_frag(bv1, b1h, b1l);
      split_frag(bv2, b2h, b2l);

      acc1 = wmma_bf16(ah, b1h, acc1);
      acc1 = wmma_bf16(ah, b1l, acc1);
      acc1 = wmma_bf16(al, b1h, acc1);
      acc2 = wmma_bf16(ah, b2h, acc2);
      acc2 = wmma_bf16(ah, b2l, acc2);
      acc2 = wmma_bf16(al, b2h, acc2);
    }

    // ---- SwiGLU -> ActC ----
    __syncthreads();  // previous chunk's GEMM2 reads are done
    const int cl = wave * 16 + r;  // column within chunk [0,128)
#pragma unroll
    for (int i = 0; i < 8; ++i) {
      float h1 = acc1[i];
      float h2 = acc2[i];
      ActC[i + 8 * half][cl] = h1 * (h2 / (1.0f + expf(-h2)));  // h1*silu(h2)
    }
    __syncthreads();

    // ---- GEMM2 partial: K slice of 128 in chunks of 32 ----
    for (int kc = 0; kc < 128; kc += 32) {
      float av[16];
      const int ka = kc + 16 * half;
#pragma unroll
      for (int j = 0; j < 8; ++j) av[j] = ActC[r][ka + j];
#pragma unroll
      for (int j = 0; j < 8; ++j) av[8 + j] = ActC[r][ka + 8 + j];
      v16bf ah, al;
      split_frag(av, ah, al);

      const int kb = fc + kc + 16 * half;
      const float* bp = w2e + (size_t)kb * D_DIM + n0 + r;
#pragma unroll
      for (int t = 0; t < 4; ++t) {
        float bv[16];
#pragma unroll
        for (int j = 0; j < 16; ++j) bv[j] = bp[(size_t)j * D_DIM + t * 16];
        v16bf bh, bl;
        split_frag(bv, bh, bl);
        oacc[t] = wmma_bf16(ah, bh, oacc[t]);
        oacc[t] = wmma_bf16(ah, bl, oacc[t]);
        oacc[t] = wmma_bf16(al, bh, oacc[t]);
      }
    }
  }

  // ---- scale by sel_w and scatter-add ----
#pragma unroll
  for (int i = 0; i < 8; ++i) {
    const int row = i + 8 * half;
    const int tok = sel_idx[e * CAP + m0 + row];
    const float wv = sel_w[e * CAP + m0 + row];
    float* dst = out + (size_t)tok * D_DIM + n0 + r;
    atomicAdd(dst +  0, oacc[0][i] * wv);
    atomicAdd(dst + 16, oacc[1][i] * wv);
    atomicAdd(dst + 32, oacc[2][i] * wv);
    atomicAdd(dst + 48, oacc[3][i] * wv);
  }
}

// ---------------------------------------------------------------- launch
extern "C" void kernel_launch(void* const* d_in, const int* in_sizes, int n_in,
                              void* d_out, int out_size, void* d_ws, size_t ws_size,
                              hipStream_t stream) {
  const float* x  = (const float*)d_in[0];   // [8,4096,512]
  const float* wg = (const float*)d_in[1];   // [512,16]
  const float* w1 = (const float*)d_in[2];   // [16,512,2048]
  const float* w2 = (const float*)d_in[3];   // [16,1024,512]
  float* out = (float*)d_out;                // [T*D] combined + [1] aux

  // workspace layout (~2.4 MB)
  float* w_et   = (float*)d_ws;                           // E*T
  int*   selidx = (int*)(w_et + (size_t)E_NUM * T_TOK);   // E*CAP
  float* selw   = (float*)(selidx + E_NUM * CAP);         // E*CAP
  float* esum   = selw + E_NUM * CAP;                     // E

  const int combined_n = T_TOK * D_DIM;  // out_size == combined_n + 1

  moe_zero_kernel<<<(out_size + 255) / 256, 256, 0, stream>>>(out, out_size, esum);

  moe_gate_kernel<<<T_TOK / 16, 256, 0, stream>>>(x, wg, w_et, esum);

  moe_select_kernel<<<E_NUM, 256, 0, stream>>>(w_et, selidx, selw);

  moe_aux_kernel<<<1, 32, 0, stream>>>(esum, out + combined_n);

  dim3 grid(CAP / 16, E_NUM);
  moe_expert_kernel<<<grid, 256, 0, stream>>>(x, w1, w2, selidx, selw, out);
}
